// DocumentGAT_48198122996252
// MI455X (gfx1250) — matrix operand order repaired and checked by reading, not compiled, not run on hardware
//
#include <hip/hip_runtime.h>

#define NN     4096
#define HEADS  4
#define DHEAD  64
#define FINN   20
#define F1     256   // HEADS*DHEAD
#define SPLITS 4
#define JCHUNK (NN / SPLITS)

typedef __attribute__((ext_vector_type(16))) __bf16 v16bf;
typedef __attribute__((ext_vector_type(8)))  float  v8f;

union Frag16 { v16bf v; unsigned short s[16]; uint4 q[2]; };
union Acc8   { v8f v; float f[8]; };

struct Tile {                 // one 16x32 attention tile's staged inputs
  Frag16 B[4];                // 4 V-tile B fragments (16x32 bf16 each)
  float4 d[4];                // f_d values for this lane's K pattern
  unsigned mw;                // adjacency mask word
};

__device__ __forceinline__ int laneid() { return (int)(threadIdx.x & 31u); }

__device__ __forceinline__ float shflx(float v, int m) {
  return __int_as_float(__builtin_amdgcn_ds_bpermute(((laneid() ^ m) & 31) << 2,
                                                     __float_as_int(v)));
}
__device__ __forceinline__ unsigned short f2bf(float f) {
  unsigned u = __float_as_uint(f);
  u += 0x7fffu + ((u >> 16) & 1u);           // RNE to bf16
  return (unsigned short)(u >> 16);
}
__device__ __forceinline__ float lrelu(float s) {
  return fmaxf(s, 0.f) + 0.2f * fminf(s, 0.f);
}
__device__ __forceinline__ float elu(float o) {
  return (o > 0.f) ? o : (__expf(o) - 1.f);
}

// ---------------------------------------------------------------------------
// Kernel 1: pack adjacency (int32 0/1) into bitmask rows of 128 uint32 words.
// ---------------------------------------------------------------------------
__global__ void __launch_bounds__(256) k_pack_adj(const int* __restrict__ adj,
                                                  unsigned* __restrict__ bm) {
  int wid  = (int)((blockIdx.x * blockDim.x + threadIdx.x) >> 5);
  int lane = laneid();
  int row  = wid >> 7;
  int word = wid & 127;
  int v = adj[(size_t)row * NN + word * 32 + lane] > 0;
#if __has_builtin(__builtin_amdgcn_ballot_w32)
  unsigned mask = __builtin_amdgcn_ballot_w32(v != 0);
  if (lane == 0) bm[(size_t)row * 128 + word] = mask;
#else
  __shared__ unsigned tmp[8];
  if ((threadIdx.x & 31u) == 0) tmp[threadIdx.x >> 5] = 0;
  __syncthreads();
  atomicOr(&tmp[threadIdx.x >> 5], ((unsigned)v) << lane);
  __syncthreads();
  if (lane == 0) bm[(size_t)row * 128 + word] = tmp[threadIdx.x >> 5];
#endif
}

// ---------------------------------------------------------------------------
// Kernel 2: layer-0 projection (K=20, fp32 VALU): Wh0 = x @ W0 per head.
// ---------------------------------------------------------------------------
__global__ void __launch_bounds__(64) k_proj0(const float* __restrict__ x,
                                              const float* __restrict__ W0,
                                              const float* __restrict__ a0,
                                              unsigned short* __restrict__ Vt0,
                                              float* __restrict__ fs0,
                                              float* __restrict__ fd0) {
  int n = (int)blockIdx.x, head = (int)blockIdx.y, o = (int)threadIdx.x;
  const float* xr = x + (size_t)n * FINN;
  const float* w  = W0 + ((size_t)head * FINN) * DHEAD + o;
  float acc = 0.f;
#pragma unroll
  for (int k = 0; k < FINN; ++k) acc += xr[k] * w[(size_t)k * DHEAD];
  Vt0[((size_t)head * DHEAD + o) * NN + n] = f2bf(acc);
  float vs = acc * a0[head * 2 * DHEAD + o];
  float vd = acc * a0[head * 2 * DHEAD + DHEAD + o];
#pragma unroll
  for (int m = 1; m < 32; m <<= 1) { vs += shflx(vs, m); vd += shflx(vd, m); }
  __shared__ float red[4];
  int wv = (int)(threadIdx.x >> 5);
  if ((threadIdx.x & 31u) == 0) { red[wv * 2] = vs; red[wv * 2 + 1] = vd; }
  __syncthreads();
  if (threadIdx.x == 0) {
    fs0[(size_t)head * NN + n] = red[0] + red[2];
    fd0[(size_t)head * NN + n] = red[1] + red[3];
  }
}

// ---------------------------------------------------------------------------
// Kernel 3: convert W1 fp32 [H][256][64] -> W1t bf16 col-major [H][64][256]
// ---------------------------------------------------------------------------
__global__ void __launch_bounds__(256) k_convW1(const float* __restrict__ W1,
                                                unsigned short* __restrict__ W1t) {
  int tid = (int)(blockIdx.x * blockDim.x + threadIdx.x); // 65536
  int h = tid >> 14, rem = tid & 16383, k = rem >> 6, o = rem & 63;
  W1t[((size_t)h * DHEAD + o) * F1 + k] = f2bf(W1[tid]);
}

// ---------------------------------------------------------------------------
// Kernel 4: exact softmax row max. LeakyReLU is monotone, so
//   max_j e_ij = LReLU(f_s[i] + max_{j in nbr(i)} f_d[j]),  or NEG if no nbrs.
// ---------------------------------------------------------------------------
__global__ void __launch_bounds__(32) k_rowmax(const float* __restrict__ fs,
                                               const float* __restrict__ fd,
                                               const unsigned* __restrict__ bm,
                                               float* __restrict__ Mrow) {
  int row = (int)blockIdx.x, head = (int)blockIdx.y, lane = laneid();
  const float* fdh = fd + (size_t)head * NN;
  const unsigned* bmr = bm + (size_t)row * 128;
  float mx = -INFINITY;
#pragma unroll
  for (int it = 0; it < 4; ++it) {
    int w = lane + it * 32;
    unsigned mw = bmr[w];
#pragma unroll
    for (int c4 = 0; c4 < 8; ++c4) {
      float4 d = *(const float4*)(fdh + w * 32 + c4 * 4);
      float v0 = ((mw >> (c4 * 4 + 0)) & 1u) ? d.x : -INFINITY;
      float v1 = ((mw >> (c4 * 4 + 1)) & 1u) ? d.y : -INFINITY;
      float v2 = ((mw >> (c4 * 4 + 2)) & 1u) ? d.z : -INFINITY;
      float v3 = ((mw >> (c4 * 4 + 3)) & 1u) ? d.w : -INFINITY;
      mx = fmaxf(mx, fmaxf(fmaxf(v0, v1), fmaxf(v2, v3)));
    }
  }
#pragma unroll
  for (int m = 1; m < 32; m <<= 1) mx = fmaxf(mx, shflx(mx, m));
  if (lane == 0) {
    float M = (mx == -INFINITY) ? -9.0e15f
                                : lrelu(fs[(size_t)head * NN + row] + mx);
    Mrow[(size_t)head * NN + row] = M;
  }
}

// ---------------------------------------------------------------------------
// Kernel 5: fused GAT attention, exact-max softmax, column-split partials.
// 2-stage software pipeline: tile j+32 is fully staged (mask, f_d, 8x B128)
// before tile j is consumed (exp chain + 4 WMMAs), so the WMMA waits are
// satisfied by a whole tile of compute. Ping-pong buffers keep both tiles
// live so the register allocator cannot re-serialize the loads.
// ---------------------------------------------------------------------------
__global__ void __launch_bounds__(32) k_attn(const float* __restrict__ fs,
                                             const float* __restrict__ fd,
                                             const float* __restrict__ Mrow,
                                             const unsigned short* __restrict__ Vt,
                                             const unsigned* __restrict__ bm,
                                             float* __restrict__ Opart,
                                             float* __restrict__ Lpart) {
  const int head = (int)blockIdx.y;
  const int sp   = (int)blockIdx.z;
  const int i0   = (int)blockIdx.x * 16;
  const int lane = laneid();
  const int m    = lane & 15;
  const int half = lane >> 4;
  const int kb   = half * 8;            // A-frag K base for this lane
  const int row  = i0 + m;

  const float  fsi = fs[(size_t)head * NN + row];
  const float  Mr  = Mrow[(size_t)head * NN + row];
  const float  pz  = __expf(-9.0e15f - Mr);   // 1 for isolated rows, else 0
  const float* fdh = fd + (size_t)head * NN;
  const unsigned short* V = Vt + (size_t)head * DHEAD * NN;
  const unsigned* bmr = bm + (size_t)row * 128;

  Acc8 acc[4];
#pragma unroll
  for (int t = 0; t < 4; ++t)
#pragma unroll
    for (int r = 0; r < 8; ++r) acc[t].f[r] = 0.f;
  float lsum = 0.f;

  auto stage = [&](Tile& T, int j0) {
    T.mw   = bmr[j0 >> 5];
    T.d[0] = *(const float4*)(fdh + j0 + kb);
    T.d[1] = *(const float4*)(fdh + j0 + kb + 4);
    T.d[2] = *(const float4*)(fdh + j0 + kb + 16);
    T.d[3] = *(const float4*)(fdh + j0 + kb + 20);
#pragma unroll
    for (int t = 0; t < 4; ++t) {
      const unsigned short* bp = V + ((size_t)(t * 16 + m)) * NN + j0 + half * 8;
      T.B[t].q[0] = *(const uint4*)bp;
      T.B[t].q[1] = *(const uint4*)(bp + 16);
    }
  };

  auto consume = [&](Tile& T) {
    float fdv[16] = {T.d[0].x, T.d[0].y, T.d[0].z, T.d[0].w,
                     T.d[1].x, T.d[1].y, T.d[1].z, T.d[1].w,
                     T.d[2].x, T.d[2].y, T.d[2].z, T.d[2].w,
                     T.d[3].x, T.d[3].y, T.d[3].z, T.d[3].w};
    Frag16 A;
#pragma unroll
    for (int i = 0; i < 16; ++i) {
      int k = kb + i + ((i < 8) ? 0 : 8);           // A-layout K index
      float bit = (float)((T.mw >> k) & 1u);
      float arg = fminf(lrelu(fsi + fdv[i]) - Mr, 0.f);
      float pu  = __expf(arg);
      float p   = fmaf(bit, pu - pz, pz);           // bit? pu : pz
      lsum += p;
      A.s[i] = f2bf(p);
    }
#pragma unroll
    for (int t = 0; t < 4; ++t)
      acc[t].v = __builtin_amdgcn_wmma_f32_16x16x32_bf16(
          false, A.v, false, T.B[t].v, (short)0, acc[t].v, false, false);
  };

  const int jbeg = sp * JCHUNK;
  const int jend = jbeg + JCHUNK;
  Tile T0, T1;
  stage(T0, jbeg);
  // JCHUNK/32 = 32 tiles per split -> 16 double iterations, no remainder.
  for (int j0 = jbeg; j0 < jend; j0 += 64) {
    int j1 = j0 + 32;                                 // always < jend
    stage(T1, j1);
    consume(T0);
    int j2 = (j0 + 64 < jend) ? (j0 + 64) : j1;       // clamp final prefetch
    stage(T0, j2);
    consume(T1);
  }

  lsum += shflx(lsum, 16);                            // full row partial sum
  if (half == 0)
    Lpart[((size_t)sp * HEADS + head) * NN + row] = lsum;
#pragma unroll
  for (int t = 0; t < 4; ++t)
#pragma unroll
    for (int r = 0; r < 8; ++r) {
      int rr = i0 + half * 8 + r;
      Opart[(((size_t)sp * HEADS + head) * NN + rr) * DHEAD + t * 16 + m] =
          acc[t].f[r];
    }
}

// ---------------------------------------------------------------------------
// Kernel 6a: merge splits for layer 0 -> ELU -> bf16 concat h [N][256]
// ---------------------------------------------------------------------------
__global__ void __launch_bounds__(256) k_combine0(const float* __restrict__ Opart,
                                                  const float* __restrict__ Lpart,
                                                  unsigned short* __restrict__ hb) {
  int tid = (int)(blockIdx.x * blockDim.x + threadIdx.x); // H*N*64
  int h = tid >> 18, rem = tid & ((NN * DHEAD) - 1);
  int row = rem >> 6, col = rem & 63;
  float a = 0.f, l = 0.f;
#pragma unroll
  for (int s = 0; s < SPLITS; ++s) {
    a += Opart[(((size_t)s * HEADS + h) * NN + row) * DHEAD + col];
    l += Lpart[((size_t)s * HEADS + h) * NN + row];
  }
  hb[(size_t)row * F1 + h * DHEAD + col] = f2bf(elu(a / l));
}

// ---------------------------------------------------------------------------
// Kernel 6b: merge splits for layer 1 -> ELU -> mean over heads -> out
// ---------------------------------------------------------------------------
__global__ void __launch_bounds__(256) k_combine1(const float* __restrict__ Opart,
                                                  const float* __restrict__ Lpart,
                                                  float* __restrict__ out) {
  int tid = (int)(blockIdx.x * blockDim.x + threadIdx.x); // N*64
  int row = tid >> 6, col = tid & 63;
  float sum = 0.f;
#pragma unroll
  for (int h = 0; h < HEADS; ++h) {
    float a = 0.f, l = 0.f;
#pragma unroll
    for (int s = 0; s < SPLITS; ++s) {
      a += Opart[(((size_t)s * HEADS + h) * NN + row) * DHEAD + col];
      l += Lpart[((size_t)s * HEADS + h) * NN + row];
    }
    sum += elu(a / l);
  }
  out[tid] = 0.25f * sum;
}

// ---------------------------------------------------------------------------
// Kernel 7: layer-1 projection via bf16 WMMA: Wh1 = h(bf16) @ W1t, K=256.
// ---------------------------------------------------------------------------
__global__ void __launch_bounds__(32) k_proj1(const unsigned short* __restrict__ hb,
                                              const unsigned short* __restrict__ W1t,
                                              const float* __restrict__ a1,
                                              unsigned short* __restrict__ Vt1,
                                              float* __restrict__ fs1,
                                              float* __restrict__ fd1) {
  const int head = (int)blockIdx.y;
  const int i0   = (int)blockIdx.x * 16;
  const int lane = laneid();
  const int m    = lane & 15;
  const int half = lane >> 4;
  const int arow = i0 + m;

  Acc8 acc[4];
#pragma unroll
  for (int t = 0; t < 4; ++t)
#pragma unroll
    for (int r = 0; r < 8; ++r) acc[t].f[r] = 0.f;

#pragma unroll
  for (int k0 = 0; k0 < F1; k0 += 32) {
    Frag16 A, B[4];
    const unsigned short* ap = hb + (size_t)arow * F1 + k0 + half * 8;
    A.q[0] = *(const uint4*)ap;          // K = k0+kb .. +7
    A.q[1] = *(const uint4*)(ap + 16);   // K = k0+kb+16 .. +23
#pragma unroll
    for (int t = 0; t < 4; ++t) {
      const unsigned short* bp =
          W1t + ((size_t)head * DHEAD + t * 16 + m) * F1 + k0 + half * 8;
      B[t].q[0] = *(const uint4*)bp;
      B[t].q[1] = *(const uint4*)(bp + 16);
    }
#pragma unroll
    for (int t = 0; t < 4; ++t)
      acc[t].v = __builtin_amdgcn_wmma_f32_16x16x32_bf16(
          false, A.v, false, B[t].v, (short)0, acc[t].v, false, false);
  }

  // attention logits contributions + transposed bf16 store
  float ps[8], pd[8];
#pragma unroll
  for (int r = 0; r < 8; ++r) { ps[r] = 0.f; pd[r] = 0.f; }
#pragma unroll
  for (int t = 0; t < 4; ++t) {
    float as = a1[head * 2 * DHEAD + t * 16 + m];
    float ad = a1[head * 2 * DHEAD + DHEAD + t * 16 + m];
#pragma unroll
    for (int r = 0; r < 8; ++r) {
      ps[r] += acc[t].f[r] * as;
      pd[r] += acc[t].f[r] * ad;
    }
  }
#pragma unroll
  for (int msk = 1; msk < 16; msk <<= 1)
#pragma unroll
    for (int r = 0; r < 8; ++r) { ps[r] += shflx(ps[r], msk); pd[r] += shflx(pd[r], msk); }

  if (m == 0) {
#pragma unroll
    for (int r = 0; r < 8; ++r) {
      int rr = i0 + half * 8 + r;
      fs1[(size_t)head * NN + rr] = ps[r];
      fd1[(size_t)head * NN + rr] = pd[r];
    }
  }
#pragma unroll
  for (int t = 0; t < 4; ++t)
#pragma unroll
    for (int r = 0; r < 8; ++r) {
      int rr = i0 + half * 8 + r;
      Vt1[((size_t)head * DHEAD + t * 16 + m) * NN + rr] = f2bf(acc[t].f[r]);
    }
}

// ---------------------------------------------------------------------------
extern "C" void kernel_launch(void* const* d_in, const int* in_sizes, int n_in,
                              void* d_out, int out_size, void* d_ws, size_t ws_size,
                              hipStream_t stream) {
  (void)in_sizes; (void)n_in; (void)out_size; (void)ws_size;
  const float* x   = (const float*)d_in[0];   // [4096,20]
  const int*   adj = (const int*)  d_in[1];   // [4096,4096]
  const float* W0  = (const float*)d_in[2];   // [4,20,64]
  const float* a0  = (const float*)d_in[3];   // [4,128]
  const float* W1  = (const float*)d_in[4];   // [4,256,64]
  const float* a1  = (const float*)d_in[5];   // [4,128]
  float* out = (float*)d_out;                 // [4096,64]

  char* base = (char*)d_ws;
  size_t off = 0;
  auto take = [&](size_t bytes) -> void* {
    void* p = base + off;
    off += (bytes + 255) & ~(size_t)255;
    return p;
  };
  unsigned*       bm  = (unsigned*)      take((size_t)NN * 128 * 4);          // 2 MB
  unsigned short* Vt0 = (unsigned short*)take((size_t)HEADS * DHEAD * NN * 2);
  float*          fs0 = (float*)         take((size_t)HEADS * NN * 4);
  float*          fd0 = (float*)         take((size_t)HEADS * NN * 4);
  unsigned short* hb  = (unsigned short*)take((size_t)NN * F1 * 2);
  unsigned short* W1t = (unsigned short*)take((size_t)HEADS * DHEAD * F1 * 2);
  unsigned short* Vt1 = (unsigned short*)take((size_t)HEADS * DHEAD * NN * 2);
  float*          fs1 = (float*)         take((size_t)HEADS * NN * 4);
  float*          fd1 = (float*)         take((size_t)HEADS * NN * 4);
  float*          M0  = (float*)         take((size_t)HEADS * NN * 4);
  float*          M1  = (float*)         take((size_t)HEADS * NN * 4);
  float*          Opart = (float*)       take((size_t)SPLITS * HEADS * NN * DHEAD * 4); // 16 MB
  float*          Lpart = (float*)       take((size_t)SPLITS * HEADS * NN * 4);

  // 1) pack adjacency -> bitmask
  k_pack_adj<<<(NN * 128) / 8, 256, 0, stream>>>(adj, bm);
  // 2) layer-0 projection + logits
  k_proj0<<<dim3(NN, HEADS), 64, 0, stream>>>(x, W0, a0, Vt0, fs0, fd0);
  // 3) W1 -> bf16 col-major
  k_convW1<<<(HEADS * F1 * DHEAD) / 256, 256, 0, stream>>>(W1, W1t);
  // 4) exact row maxima, layer 0
  k_rowmax<<<dim3(NN, HEADS), 32, 0, stream>>>(fs0, fd0, bm, M0);
  // 5) layer-0 fused attention (partials over 4 column splits)
  k_attn<<<dim3(NN / 16, HEADS, SPLITS), 32, 0, stream>>>(fs0, fd0, M0, Vt0, bm,
                                                          Opart, Lpart);
  // 6) merge -> ELU -> bf16 concat h
  k_combine0<<<(HEADS * NN * DHEAD) / 256, 256, 0, stream>>>(Opart, Lpart, hb);
  // 7) layer-1 projection (WMMA) + logits
  k_proj1<<<dim3(NN / 16, HEADS), 32, 0, stream>>>(hb, W1t, a1, Vt1, fs1, fd1);
  // 8) exact row maxima, layer 1
  k_rowmax<<<dim3(NN, HEADS), 32, 0, stream>>>(fs1, fd1, bm, M1);
  // 9) layer-1 fused attention (partials)
  k_attn<<<dim3(NN / 16, HEADS, SPLITS), 32, 0, stream>>>(fs1, fd1, M1, Vt1, bm,
                                                          Opart, Lpart);
  // 10) merge -> ELU -> head mean -> out
  k_combine1<<<(NN * DHEAD) / 256, 256, 0, stream>>>(Opart, Lpart, out);
}
